// FrameTransformerEncoder_76733885710662
// MI455X (gfx1250) — compile-verified, matrix operand order checked by reading, not run on hardware
//
#include <hip/hip_runtime.h>
#include <hip/hip_bf16.h>

// ---------------------------------------------------------------------------
// FrameTransformerEncoder forward for MI455X (gfx1250, wave32, WMMA bf16 path)
// Round 2: async global->LDS staging (ASYNCcnt) + double-buffered K pipeline.
// ---------------------------------------------------------------------------

#define DEVFN static __device__ __forceinline__

constexpr int Bd = 2, Cd = 4, Fd = 256, Wd = 1024;
constexpr int NHd = 8, HDd = 32, EXPd = 4, FEd = Fd * EXPd;      // FE = 1024
constexpr long NEL = (long)Bd * Cd * Fd * Wd;                    // 2,097,152

typedef __attribute__((ext_vector_type(16))) __bf16       v16bf;
typedef __attribute__((ext_vector_type(8)))  float        v8f;
typedef __attribute__((ext_vector_type(4)))  unsigned int uint4v;

union Frag16 {
    v16bf          v;
    uint4v         q[2];
    unsigned short s[16];
};

DEVFN unsigned short f2bf(float f) {
    unsigned int u = __float_as_uint(f);
    u += 0x7FFFu + ((u >> 16) & 1u);          // round-to-nearest-even
    return (unsigned short)(u >> 16);
}
DEVFN float bf2f(unsigned short h) {
    return __uint_as_float(((unsigned int)h) << 16);
}

// --- CDNA5 async memory->LDS DMA (tracked by ASYNCcnt) ---------------------
DEVFN void async_copy_b128(void* lds_dst, const void* gsrc) {
    unsigned int ldsa = (unsigned int)(unsigned long long)lds_dst;
    asm volatile("global_load_async_to_lds_b128 %0, %1, off"
                 :: "v"(ldsa), "v"((unsigned long long)gsrc)
                 : "memory");
}
DEVFN void wait_async0() {
    asm volatile("s_wait_asynccnt 0x0" ::: "memory");
}

// ---------------------------------------------------------------------------
// float32 -> bf16 conversion (weights)
// ---------------------------------------------------------------------------
__global__ void k_f32_to_bf16(const float* __restrict__ in,
                              unsigned short* __restrict__ out, int n) {
    int i = blockIdx.x * 256 + threadIdx.x;
    if (i < n) out[i] = f2bf(in[i]);
}

// ---------------------------------------------------------------------------
// LayerNorm over F axis per (b,c,w) column; one wave32 per column.
// x f32 [B,C,F,W] -> out bf16 [B,C,F,W]
// ---------------------------------------------------------------------------
__global__ void k_layernorm_bf16(const float* __restrict__ x,
                                 const float* __restrict__ g,
                                 const float* __restrict__ be,
                                 unsigned short* __restrict__ out) {
    const int lane = threadIdx.x & 31;
    const long wv  = (long)blockIdx.x * (blockDim.x >> 5) + (threadIdx.x >> 5);
    const int w  = (int)(wv % Wd);
    const int bc = (int)(wv / Wd);
    const int c  = bc % Cd;

    const float* col = x + (long)bc * Fd * Wd + w;
    float vals[Fd / 32];
    float s = 0.f, s2 = 0.f;
    #pragma unroll
    for (int j = 0; j < Fd / 32; ++j) {
        float v = col[(lane + 32 * j) * Wd];
        vals[j] = v; s += v; s2 += v * v;
    }
    #pragma unroll
    for (int o = 16; o > 0; o >>= 1) {
        s  += __shfl_xor(s,  o, 32);
        s2 += __shfl_xor(s2, o, 32);
    }
    const float mu  = s * (1.0f / Fd);
    const float var = s2 * (1.0f / Fd) - mu * mu;
    const float rs  = rsqrtf(var + 1e-8f);

    unsigned short* oc = out + (long)bc * Fd * Wd + w;
    #pragma unroll
    for (int j = 0; j < Fd / 32; ++j) {
        int f = lane + 32 * j;
        float v = (vals[j] - mu) * rs * g[c * Fd + f] + be[c * Fd + f];
        oc[f * Wd] = f2bf(v);
    }
}

// ---------------------------------------------------------------------------
// conv3x3 SAME, no bias:  out[b,oc,f,w] = sum_{ic,df,dw} wgt * xn[b,ic,f+df,w+dw]
// ---------------------------------------------------------------------------
__global__ void k_conv3x3_bf16(const unsigned short* __restrict__ xn,
                               const float* __restrict__ wgt,
                               unsigned short* __restrict__ out) {
    long i = (long)blockIdx.x * 256 + threadIdx.x;     // over B*C*F*W
    const int w = (int)(i % Wd);
    long t = i / Wd;
    const int f  = (int)(t % Fd); t /= Fd;
    const int oc = (int)(t % Cd);
    const int b  = (int)(t / Cd);

    float acc = 0.f;
    for (int ic = 0; ic < Cd; ++ic) {
        const unsigned short* xp = xn + (long)(b * Cd + ic) * Fd * Wd;
        const float* wp = wgt + (oc * Cd + ic) * 9;
        #pragma unroll
        for (int df = -1; df <= 1; ++df) {
            int ff = f + df;
            if (ff < 0 || ff >= Fd) continue;
            #pragma unroll
            for (int dw = -1; dw <= 1; ++dw) {
                int ww = w + dw;
                if (ww < 0 || ww >= Wd) continue;
                acc += wp[(df + 1) * 3 + (dw + 1)] * bf2f(xp[(long)ff * Wd + ww]);
            }
        }
    }
    out[i] = f2bf(acc);
}

// ---------------------------------------------------------------------------
// Depthwise channel mix: out[b,oc,:,:] = sum_ic dw[oc,ic] * in[b,ic,:,:]
// ---------------------------------------------------------------------------
__global__ void k_dwmix_bf16(const unsigned short* __restrict__ in,
                             const float* __restrict__ dw,
                             unsigned short* __restrict__ out) {
    long i = (long)blockIdx.x * 256 + threadIdx.x;     // over B*C*F*W
    const long fw = i % ((long)Fd * Wd);
    long t = i / ((long)Fd * Wd);
    const int oc = (int)(t % Cd);
    const int b  = (int)(t / Cd);
    float acc = 0.f;
    #pragma unroll
    for (int ic = 0; ic < Cd; ++ic)
        acc += dw[oc * Cd + ic] * bf2f(in[(long)(b * Cd + ic) * Fd * Wd + fw]);
    out[i] = f2bf(acc);
}

// ---------------------------------------------------------------------------
// Row softmax over 1024 entries; one wave32 per row.
// ---------------------------------------------------------------------------
__global__ void k_softmax_rows(const float* __restrict__ qk,
                               unsigned short* __restrict__ P) {
    const int lane = threadIdx.x & 31;
    const long row = (long)blockIdx.x * (blockDim.x >> 5) + (threadIdx.x >> 5);
    const float* rp = qk + row * Wd;
    float v[Wd / 32];
    float mx = -3.0e38f;
    #pragma unroll
    for (int j = 0; j < Wd / 32; ++j) {
        v[j] = rp[lane + 32 * j];
        mx = fmaxf(mx, v[j]);
    }
    #pragma unroll
    for (int o = 16; o > 0; o >>= 1) mx = fmaxf(mx, __shfl_xor(mx, o, 32));
    float s = 0.f;
    #pragma unroll
    for (int j = 0; j < Wd / 32; ++j) {
        v[j] = __expf(v[j] - mx);
        s += v[j];
    }
    #pragma unroll
    for (int o = 16; o > 0; o >>= 1) s += __shfl_xor(s, o, 32);
    const float inv = 1.0f / s;
    unsigned short* op = P + row * Wd;
    #pragma unroll
    for (int j = 0; j < Wd / 32; ++j) op[lane + 32 * j] = f2bf(v[j] * inv);
}

// ---------------------------------------------------------------------------
// Generic batched bf16 WMMA GEMM:  C[z] = alpha * A[z%aMod] * B[z] (+ R[z])
//   TRANSA: A stored [K][M]; else [M][K].  TRANSB: B stored [N][K]; else [K][N].
//   TRANSC: C stored [N][M]; else [M][N]   (R shares C's geometry)
//   EPI: 0 f32, 1 f32+residual, 2 bf16, 3 bf16 relu(x)^2
// Double-buffered LDS staging: contiguous paths use async global->LDS DMA
// (ASYNCcnt); transposed paths use sync load + LDS scatter. One barrier/K-step.
// ---------------------------------------------------------------------------
template <int WAVES_M, int WAVES_N, int WN,
          bool TRANSA, bool TRANSB, bool TRANSC, int EPI>
__global__ __launch_bounds__(32 * WAVES_M * WAVES_N)
void k_gemm_wmma_bf16(const unsigned short* __restrict__ A, long sA, int aMod, int lda,
                      const unsigned short* __restrict__ B, long sB, int ldb,
                      void* __restrict__ Cout, long sC, int ldc,
                      const float* __restrict__ R,
                      int K, float alpha) {
    constexpr int KS  = 32;
    constexpr int BM  = WAVES_M * 16;
    constexpr int BN  = WAVES_N * WN * 16;
    constexpr int LDP = 40;                 // LDS row stride (elems); 80 B (16B mult)
    constexpr int NT  = 32 * WAVES_M * WAVES_N;

    __shared__ __align__(16) unsigned short lA[2][BM * LDP];   // [m][k]
    __shared__ __align__(16) unsigned short lB[2][BN * LDP];   // [n][k]

    const int tid  = threadIdx.x;
    const int lane = tid & 31;
    const int wave = tid >> 5;
    const int wm   = wave % WAVES_M;
    const int wn   = wave / WAVES_M;
    const int m0   = blockIdx.y * BM;
    const int n0   = blockIdx.x * BN;
    const int z    = blockIdx.z;

    const unsigned short* Ab = A + (long)(z % aMod) * sA;
    const unsigned short* Bb = B + (long)z * sB;

    v8f acc[WN];
    #pragma unroll
    for (int t = 0; t < WN; ++t)
        #pragma unroll
        for (int e = 0; e < 8; ++e) acc[t][e] = 0.f;

    const int half = lane >> 4;
    const int r    = lane & 15;

    // ---- tile staging (A and B) into buffer `buf` for K-offset k0 ----
    auto stage = [&](int k0, int buf) {
        if (!TRANSA) {
            for (int idx = tid; idx < BM * (KS / 8); idx += NT) {
                int m  = idx >> 2;
                int kc = (idx & 3) * 8;
                async_copy_b128(&lA[buf][m * LDP + kc],
                                Ab + (long)(m0 + m) * lda + k0 + kc);
            }
        } else {
            for (int idx = tid; idx < KS * (BM / 8); idx += NT) {
                int k  = idx / (BM / 8);
                int mc = (idx % (BM / 8)) * 8;
                uint4v d = *(const uint4v*)(Ab + (long)(k0 + k) * lda + m0 + mc);
                const unsigned short* ds = (const unsigned short*)&d;
                #pragma unroll
                for (int j = 0; j < 8; ++j) lA[buf][(mc + j) * LDP + k] = ds[j];
            }
        }
        if (!TRANSB) {
            for (int idx = tid; idx < KS * (BN / 8); idx += NT) {
                int k  = idx / (BN / 8);
                int nc = (idx % (BN / 8)) * 8;
                uint4v d = *(const uint4v*)(Bb + (long)(k0 + k) * ldb + n0 + nc);
                const unsigned short* ds = (const unsigned short*)&d;
                #pragma unroll
                for (int j = 0; j < 8; ++j) lB[buf][(nc + j) * LDP + k] = ds[j];
            }
        } else {
            for (int idx = tid; idx < BN * (KS / 8); idx += NT) {
                int n  = idx >> 2;
                int kc = (idx & 3) * 8;
                async_copy_b128(&lB[buf][n * LDP + kc],
                                Bb + (long)(n0 + n) * ldb + k0 + kc);
            }
        }
    };

    const int nK = K / KS;
    stage(0, 0);                                 // prologue

    for (int i = 0; i < nK; ++i) {
        const int cur = i & 1;
        wait_async0();                           // my async copies of tile i done
        __syncthreads();                         // all waves' tile i visible;
                                                 // also fences reuse of buf `1-cur`
        if (i + 1 < nK) stage((i + 1) * KS, 1 - cur);   // overlap DMA with WMMA

        // ---- A fragment (CDNA5 16-bit A layout: row = lane&15,
        //      K = {half*8 .. half*8+7} then {16+half*8 ..}) ----
        Frag16 fa;
        {
            const int mrow = wm * 16 + r;
            fa.q[0] = *(const uint4v*)(&lA[cur][mrow * LDP + half * 8]);
            fa.q[1] = *(const uint4v*)(&lA[cur][mrow * LDP + 16 + half * 8]);
        }
        Frag16 fb[WN];
        #pragma unroll
        for (int t = 0; t < WN; ++t) {
            const int ncol = (wn * WN + t) * 16 + r;
            fb[t].q[0] = *(const uint4v*)(&lB[cur][ncol * LDP + half * 8]);
            fb[t].q[1] = *(const uint4v*)(&lB[cur][ncol * LDP + 16 + half * 8]);
        }
        #pragma unroll
        for (int t = 0; t < WN; ++t) {
            acc[t] = __builtin_amdgcn_wmma_f32_16x16x32_bf16(
                false, fa.v, false, fb[t].v, (short)0, acc[t], false, false);
        }
    }

    // ---- epilogue (CDNA5 32-bit C/D layout: vgpr j, lanes<16 -> M=j, else M=j+8;
    //      N = lane&15) ----
    #pragma unroll
    for (int t = 0; t < WN; ++t) {
        const int n = n0 + (wn * WN + t) * 16 + r;
        #pragma unroll
        for (int j = 0; j < 8; ++j) {
            const int m = m0 + wm * 16 + half * 8 + j;
            const long off = (long)z * sC +
                             (TRANSC ? ((long)n * ldc + m) : ((long)m * ldc + n));
            float v = acc[t][j] * alpha;
            if (EPI == 1) v += R[off];
            if (EPI == 0 || EPI == 1) {
                ((float*)Cout)[off] = v;
            } else if (EPI == 2) {
                ((unsigned short*)Cout)[off] = f2bf(v);
            } else {
                float rl = v > 0.f ? v : 0.f;
                ((unsigned short*)Cout)[off] = f2bf(rl * rl);   // SquaredReLU
            }
        }
    }
}

// ---------------------------------------------------------------------------
// Host launch
// ---------------------------------------------------------------------------
extern "C" void kernel_launch(void* const* d_in, const int* in_sizes, int n_in,
                              void* d_out, int out_size, void* d_ws, size_t ws_size,
                              hipStream_t stream) {
    const float* x    = (const float*)d_in[0];
    const float* n1w  = (const float*)d_in[1];
    const float* n1b  = (const float*)d_in[2];
    const float* qcw  = (const float*)d_in[3];
    const float* qpw  = (const float*)d_in[4];
    const float* kcw  = (const float*)d_in[5];
    const float* kpw  = (const float*)d_in[6];
    const float* vcw  = (const float*)d_in[7];
    const float* vpw  = (const float*)d_in[8];
    const float* odw  = (const float*)d_in[9];
    const float* opw  = (const float*)d_in[10];
    const float* n2w  = (const float*)d_in[11];
    const float* n2b  = (const float*)d_in[12];
    const float* c1dw = (const float*)d_in[13];
    const float* c1pw = (const float*)d_in[14];
    const float* c2pw = (const float*)d_in[15];

    float* h_out  = (float*)d_out;
    float* qk_out = h_out + NEL;                 // [B,C,NH,W,W] pre-softmax scores

    // ---- workspace layout ----
    char* ws = (char*)d_ws;
    size_t off = 0;
    auto take = [&](size_t bytes) { char* p = ws + off; off += bytes; return p; };
    unsigned short* xn   = (unsigned short*)take(NEL * 2);
    unsigned short* cq   = (unsigned short*)take(NEL * 2);
    unsigned short* ck   = (unsigned short*)take(NEL * 2);
    unsigned short* cv   = (unsigned short*)take(NEL * 2);
    unsigned short* qb   = (unsigned short*)take(NEL * 2);
    unsigned short* kb   = (unsigned short*)take(NEL * 2);
    unsigned short* vb   = (unsigned short*)take(NEL * 2);
    unsigned short* ab   = (unsigned short*)take(NEL * 2);
    unsigned short* am   = (unsigned short*)take(NEL * 2);
    unsigned short* hn   = (unsigned short*)take(NEL * 2);
    unsigned short* hm   = (unsigned short*)take(NEL * 2);
    unsigned short* mid  = (unsigned short*)take((long)Bd * Cd * FEd * Wd * 2);
    unsigned short* wqb  = (unsigned short*)take((long)Cd * Fd * Fd * 2);
    unsigned short* wkb  = (unsigned short*)take((long)Cd * Fd * Fd * 2);
    unsigned short* wvb  = (unsigned short*)take((long)Cd * Fd * Fd * 2);
    unsigned short* wob  = (unsigned short*)take((long)Cd * Fd * Fd * 2);
    unsigned short* c1wb = (unsigned short*)take((long)Cd * FEd * Fd * 2);
    unsigned short* c2wb = (unsigned short*)take((long)Cd * Fd * FEd * 2);
    unsigned short* P    = (unsigned short*)take((long)Bd * Cd * NHd * Wd * Wd * 2);
    (void)ws_size; (void)n_in; (void)in_sizes; (void)out_size;

    const int nPW  = Cd * Fd * Fd;     // 262144
    const int nC1W = Cd * FEd * Fd;    // 1048576

    // 1) weights -> bf16
    k_f32_to_bf16<<<(nPW + 255) / 256, 256, 0, stream>>>(qpw, wqb, nPW);
    k_f32_to_bf16<<<(nPW + 255) / 256, 256, 0, stream>>>(kpw, wkb, nPW);
    k_f32_to_bf16<<<(nPW + 255) / 256, 256, 0, stream>>>(vpw, wvb, nPW);
    k_f32_to_bf16<<<(nPW + 255) / 256, 256, 0, stream>>>(opw, wob, nPW);
    k_f32_to_bf16<<<(nC1W + 255) / 256, 256, 0, stream>>>(c1pw, c1wb, nC1W);
    k_f32_to_bf16<<<(nC1W + 255) / 256, 256, 0, stream>>>(c2pw, c2wb, nC1W);

    const int lnBlocks = (Bd * Cd * Wd) / 8;               // 1024 (8 waves/block)
    const int ewBlocks = (int)(NEL / 256);                 // 8192

    // 2) LN1
    k_layernorm_bf16<<<lnBlocks, 256, 0, stream>>>(x, n1w, n1b, xn);

    // 3) conv3x3 for q, k, v
    k_conv3x3_bf16<<<ewBlocks, 256, 0, stream>>>(xn, qcw, cq);
    k_conv3x3_bf16<<<ewBlocks, 256, 0, stream>>>(xn, kcw, ck);
    k_conv3x3_bf16<<<ewBlocks, 256, 0, stream>>>(xn, vcw, cv);

    // 4) per-channel position-wise projections: Y[F,W] = Wpw[c] * conv[F,W]
    {
        dim3 g(Wd / 128, Fd / 64, Bd * Cd);
        k_gemm_wmma_bf16<4, 2, 4, false, false, false, 2><<<g, 256, 0, stream>>>(
            wqb, (long)Fd * Fd, Cd, Fd, cq, (long)Fd * Wd, Wd,
            qb, (long)Fd * Wd, Wd, nullptr, Fd, 1.0f);
        k_gemm_wmma_bf16<4, 2, 4, false, false, false, 2><<<g, 256, 0, stream>>>(
            wkb, (long)Fd * Fd, Cd, Fd, ck, (long)Fd * Wd, Wd,
            kb, (long)Fd * Wd, Wd, nullptr, Fd, 1.0f);
        k_gemm_wmma_bf16<4, 2, 4, false, false, false, 2><<<g, 256, 0, stream>>>(
            wvb, (long)Fd * Fd, Cd, Fd, cv, (long)Fd * Wd, Wd,
            vb, (long)Fd * Wd, Wd, nullptr, Fd, 1.0f);
    }

    const int ZATT = Bd * Cd * NHd;                        // 64 attention batches

    // 5) scores: qk[z,i,j] = (1/sqrt(F)) * sum_d q[z,d,i] * k[z,d,j]
    {
        dim3 g(Wd / 128, Wd / 64, ZATT);
        k_gemm_wmma_bf16<4, 2, 4, true, false, false, 0><<<g, 256, 0, stream>>>(
            qb, (long)HDd * Wd, ZATT, Wd, kb, (long)HDd * Wd, Wd,
            qk_out, (long)Wd * Wd, Wd, nullptr, HDd, 0.0625f /* 1/sqrt(256) */);
    }

    // 6) softmax over last axis -> P (bf16)
    k_softmax_rows<<<(ZATT * Wd) / 8, 256, 0, stream>>>(qk_out, P);

    // 7) attn out: a[z,d,i] = sum_j P[z,i,j] * v[z,d,j]
    {
        dim3 g(HDd / 32, Wd / 128, ZATT);
        k_gemm_wmma_bf16<8, 1, 2, false, true, true, 2><<<g, 256, 0, stream>>>(
            P, (long)Wd * Wd, ZATT, Wd, vb, (long)HDd * Wd, Wd,
            ab, (long)HDd * Wd, Wd, nullptr, Wd, 1.0f);
    }

    // 8) output projection: dw channel mix, then per-channel pw GEMM + residual x
    k_dwmix_bf16<<<ewBlocks, 256, 0, stream>>>(ab, odw, am);
    {
        dim3 g(Wd / 128, Fd / 64, Bd * Cd);
        k_gemm_wmma_bf16<4, 2, 4, false, false, false, 1><<<g, 256, 0, stream>>>(
            wob, (long)Fd * Fd, Cd, Fd, am, (long)Fd * Wd, Wd,
            h_out, (long)Fd * Wd, Wd, x, Fd, 1.0f);
    }

    // 9) LN2 on h, dw mix, conv1 (expand) with fused SquaredReLU
    k_layernorm_bf16<<<lnBlocks, 256, 0, stream>>>(h_out, n2w, n2b, hn);
    k_dwmix_bf16<<<ewBlocks, 256, 0, stream>>>(hn, c1dw, hm);
    {
        dim3 g(Wd / 128, FEd / 64, Bd * Cd);
        k_gemm_wmma_bf16<4, 2, 4, false, false, false, 3><<<g, 256, 0, stream>>>(
            c1wb, (long)FEd * Fd, Cd, Fd, hm, (long)Fd * Wd, Wd,
            mid, (long)FEd * Wd, Wd, nullptr, Fd, 1.0f);
    }

    // 10) conv2 (contract) + residual h  (in-place on h_out: per-element RMW)
    {
        dim3 g(Wd / 128, Fd / 64, Bd * Cd);
        k_gemm_wmma_bf16<4, 2, 4, false, false, false, 1><<<g, 256, 0, stream>>>(
            c2wb, (long)Fd * FEd, Cd, FEd, mid, (long)FEd * Wd, Wd,
            h_out, (long)Fd * Wd, Wd, h_out, FEd, 1.0f);
    }
}
